// GCNSyntheticPerturb_29351806501599
// MI455X (gfx1250) — compile-verified, matrix operand order checked by reading, not run on hardware
//
#include <hip/hip_runtime.h>
#include <hip/hip_bf16.h>

// ---------------------------------------------------------------------------
// Types for CDNA5 WMMA (wave32): v_wmma_f32_16x16x32_bf16
// ---------------------------------------------------------------------------
typedef __attribute__((ext_vector_type(16))) __bf16        v16bf;
typedef __attribute__((ext_vector_type(8)))  __bf16        v8bf;
typedef __attribute__((ext_vector_type(8)))  float         v8f;

__device__ __forceinline__ __bf16 tobf(float f) { return (__bf16)f; }

// ---------------------------------------------------------------------------
// 1) zero-fill
// ---------------------------------------------------------------------------
__global__ void k_fill0(float* __restrict__ p, int n) {
  int i = blockIdx.x * blockDim.x + threadIdx.x;
  if (i < n) p[i] = 0.0f;
}

// ---------------------------------------------------------------------------
// 2) w[e] = sigmoid(P[e]); deg[col[e]] += w[e]
// ---------------------------------------------------------------------------
__global__ void k_sig_deg(const float* __restrict__ P, const int* __restrict__ col,
                          float* __restrict__ w, float* __restrict__ deg, int E) {
  int e = blockIdx.x * blockDim.x + threadIdx.x;
  if (e >= E) return;
  float s = 1.0f / (1.0f + __expf(-P[e]));
  w[e] = s;
  atomicAdd(&deg[col[e]], s);
}

// ---------------------------------------------------------------------------
// 3) dinv[i] = (deg[i]+1)^-0.5   (self-loop weight 1 => deg always > 0)
// ---------------------------------------------------------------------------
__global__ void k_dinv(float* __restrict__ deg, int N) {
  int i = blockIdx.x * blockDim.x + threadIdx.x;
  if (i >= N) return;
  float d = deg[i] + 1.0f;
  deg[i] = (d > 0.0f) ? rsqrtf(d) : 0.0f;
}

// ---------------------------------------------------------------------------
// 4) norm[e] = dinv[row]*w[e]*dinv[col]  (in place over w)
// ---------------------------------------------------------------------------
__global__ void k_norm(const int* __restrict__ row, const int* __restrict__ col,
                       const float* __restrict__ dinv, float* __restrict__ w, int E) {
  int e = blockIdx.x * blockDim.x + threadIdx.x;
  if (e >= E) return;
  w[e] = dinv[row[e]] * w[e] * dinv[col[e]];
}

// ---------------------------------------------------------------------------
// 5) Transpose + convert W [K=128 x NCOL] fp32 -> Wt [NCOL x 128] bf16.
//    Tiny (16K/8K elems); Wt stays L2-resident (192 MB L2).
// ---------------------------------------------------------------------------
template<int NCOL>
__global__ void k_wt_bf16(const float* __restrict__ W, __bf16* __restrict__ Wt) {
  int idx = blockIdx.x * blockDim.x + threadIdx.x;
  if (idx >= 128 * NCOL) return;
  int k = idx / NCOL, n = idx % NCOL;
  Wt[n * 128 + k] = tobf(W[idx]);
}

// ---------------------------------------------------------------------------
// WMMA GEMM with fused epilogue:
//   H[m,n]   = (relu?)A[m,:] @ Bt[n,:]          (bf16 x bf16 -> f32)
//   OUT[m,n] = bias[n] + dinv[m]^2 * H[m,n]     (self-loop + bias init)
//
// One block per 16-row strip. The strip is staged ONCE into LDS as bf16
// (row stride padded 128->136 elems so the 16-lane ds_load_b128 fragment
// reads hit all 64 banks conflict-free). NCOL/16 waves per block, one
// 16x16 C tile per wave; K=128 fully unrolled -> 4 v_wmma_f32_16x16x32_bf16.
//
// Fragment layout (ISA 7.12.2, 16-bit A 16x32): lane half h = lane>>4,
//   frag elem e in [0,8)  -> K = k0 + h*8 + e        (16B contiguous)
//   frag elem e in [8,16) -> K = k0 + 16 + h*8 + (e-8)
// B fragment (lane = column n) uses the identical per-lane K pattern.
// ---------------------------------------------------------------------------
template<int NCOL, bool RELU_A>
__global__ void k_gemm_wmma(const float* __restrict__ A,
                            const __bf16* __restrict__ Bt,
                            const float* __restrict__ bias,
                            const float* __restrict__ dinv,
                            float* __restrict__ H,
                            float* __restrict__ OUT, int N) {
  constexpr int K    = 128;
  constexpr int LDSP = 136;                 // padded row stride (bf16 elems)
  __shared__ __align__(16) __bf16 As[16 * LDSP];

  const int lane = threadIdx.x & 31;
  const int wave = threadIdx.x >> 5;
  const int m0   = blockIdx.x * 16;
  const int n0   = wave * 16;
  const int half = lane >> 4;               // 0: lanes 0-15, 1: lanes 16-31
  const int l15  = lane & 15;

  // ---- cooperative stage: 16x128 fp32 -> bf16 in LDS (once per block) ----
  for (int base = threadIdx.x * 8; base < 16 * K; base += blockDim.x * 8) {
    const int m = base >> 7;                // row in strip
    const int k = base & (K - 1);           // col (multiple of 8)
    int mg = m0 + m; if (mg > N - 1) mg = N - 1;   // clamp (keeps EXEC full)
    const float* src = A + (size_t)mg * K + k;
    float4 f0 = *(const float4*)(src);
    float4 f1 = *(const float4*)(src + 4);
    if (RELU_A) {
      f0.x = fmaxf(f0.x, 0.f); f0.y = fmaxf(f0.y, 0.f);
      f0.z = fmaxf(f0.z, 0.f); f0.w = fmaxf(f0.w, 0.f);
      f1.x = fmaxf(f1.x, 0.f); f1.y = fmaxf(f1.y, 0.f);
      f1.z = fmaxf(f1.z, 0.f); f1.w = fmaxf(f1.w, 0.f);
    }
    v8bf pk;
    pk[0] = tobf(f0.x); pk[1] = tobf(f0.y); pk[2] = tobf(f0.z); pk[3] = tobf(f0.w);
    pk[4] = tobf(f1.x); pk[5] = tobf(f1.y); pk[6] = tobf(f1.z); pk[7] = tobf(f1.w);
    *(v8bf*)(&As[m * LDSP + k]) = pk;       // 16B ds_store
  }
  __syncthreads();

  const __bf16* Brow = Bt + (size_t)(n0 + l15) * K;
  const __bf16* Alds = As + l15 * LDSP;

  v8f acc = {};

#pragma unroll
  for (int k0 = 0; k0 < K; k0 += 32) {
    // A fragment: two ds_load_b128 (bank-conflict-free via LDSP padding)
    const v8bf a0 = *(const v8bf*)(Alds + k0 + half * 8);
    const v8bf a1 = *(const v8bf*)(Alds + k0 + 16 + half * 8);
    const v16bf a = __builtin_shufflevector(a0, a1,
        0, 1, 2, 3, 4, 5, 6, 7, 8, 9, 10, 11, 12, 13, 14, 15);
    // B fragment: two global_load_b128 from L2-resident bf16 Wt
    const v8bf b0 = *(const v8bf*)(Brow + k0 + half * 8);
    const v8bf b1 = *(const v8bf*)(Brow + k0 + 16 + half * 8);
    const v16bf b = __builtin_shufflevector(b0, b1,
        0, 1, 2, 3, 4, 5, 6, 7, 8, 9, 10, 11, 12, 13, 14, 15);
    // (neg_a, A, neg_b, B, c_mod, C, reuse_a, reuse_b)
    acc = __builtin_amdgcn_wmma_f32_16x16x32_bf16(false, a, false, b,
                                                  (short)0, acc, false, false);
  }

  // ---- fused epilogue: H = h, OUT = bias[n] + dinv[m]^2 * h ----
  // C/D layout: lanes 0-15 -> N=lane, M=r; lanes 16-31 -> N=lane-16, M=8+r
  const int   n  = n0 + l15;
  const float bn = bias[n];
  if (m0 + 16 <= N) {                       // common branch-free path
#pragma unroll
    for (int r = 0; r < 8; ++r) {
      const int   m = m0 + half * 8 + r;
      const float d = dinv[m];
      const float h = acc[r];
      H  [(size_t)m * NCOL + n] = h;
      OUT[(size_t)m * NCOL + n] = fmaf(d * d, h, bn);
    }
  } else {                                  // rare ragged tail
#pragma unroll
    for (int r = 0; r < 8; ++r) {
      const int m = m0 + half * 8 + r;
      if (m < N) {
        const float d = dinv[m];
        const float h = acc[r];
        H  [(size_t)m * NCOL + n] = h;
        OUT[(size_t)m * NCOL + n] = fmaf(d * d, h, bn);
      }
    }
  }
}

// ---------------------------------------------------------------------------
// Edge scatter: one wave32 per edge; out[col,:] += norm[e] * h[row,:]
// Edge descriptors are wave-uniform -> force onto the scalar (SMEM) path.
// ---------------------------------------------------------------------------
template<int F>
__global__ void k_scatter(const int* __restrict__ row, const int* __restrict__ col,
                          const float* __restrict__ nrm, const float* __restrict__ h,
                          float* __restrict__ out, int E) {
  const int wid  = __builtin_amdgcn_readfirstlane(
                     (blockIdx.x * blockDim.x + threadIdx.x) >> 5);
  const int lane = threadIdx.x & 31;
  if (wid >= E) return;
  const int   r  = row[wid];
  const int   c  = col[wid];
  const float nm = nrm[wid];
  constexpr int PER = F / 32;                     // 4 (F=128) or 2 (F=64)
  const float* hp = h   + (size_t)r * F + lane * PER;
  float*       op = out + (size_t)c * F + lane * PER;
  if constexpr (PER == 4) {
    float4 v = *(const float4*)hp;
    atomicAdd(op + 0, nm * v.x);
    atomicAdd(op + 1, nm * v.y);
    atomicAdd(op + 2, nm * v.z);
    atomicAdd(op + 3, nm * v.w);
  } else {
    float2 v = *(const float2*)hp;
    atomicAdd(op + 0, nm * v.x);
    atomicAdd(op + 1, nm * v.y);
  }
}

// ---------------------------------------------------------------------------
// Host orchestration
// ---------------------------------------------------------------------------
extern "C" void kernel_launch(void* const* d_in, const int* in_sizes, int n_in,
                              void* d_out, int out_size, void* d_ws, size_t ws_size,
                              hipStream_t stream) {
  const float* x  = (const float*)d_in[0];  // [N,128]
  const int*   ei = (const int*)  d_in[1];  // [2,E]
  const float* P  = (const float*)d_in[2];  // [E]
  const float* W1 = (const float*)d_in[3];  // [128,128]
  const float* b1 = (const float*)d_in[4];  // [128]
  const float* W2 = (const float*)d_in[5];  // [128,64]
  const float* b2 = (const float*)d_in[6];  // [64]
  float* out = (float*)d_out;               // [N,64]

  constexpr int F_IN = 128, H1 = 128, H2 = 64;
  const int N = in_sizes[0] / F_IN;
  const int E = in_sizes[2];
  const int* row = ei;
  const int* col = ei + E;

  // ws partition (floats): dinv[N] | norm[E] | h[N*128] | o1[N*128] | W1t | W2t
  float* dinv = (float*)d_ws;
  float* wn   = dinv + N;
  float* h    = wn + E;
  float* o1   = h + (size_t)N * H1;
  __bf16* w1t = (__bf16*)(o1 + (size_t)N * H1);
  __bf16* w2t = w1t + F_IN * H1;

  const int T = 256;

  // --- degree / norm prep + weight transpose/convert ---
  k_fill0  <<<(N + T - 1) / T, T, 0, stream>>>(dinv, N);
  k_sig_deg<<<(E + T - 1) / T, T, 0, stream>>>(P, col, wn, dinv, E);
  k_dinv   <<<(N + T - 1) / T, T, 0, stream>>>(dinv, N);
  k_norm   <<<(E + T - 1) / T, T, 0, stream>>>(row, col, dinv, wn, E);
  k_wt_bf16<H1><<<(F_IN * H1 + T - 1) / T, T, 0, stream>>>(W1, w1t);
  k_wt_bf16<H2><<<(F_IN * H2 + T - 1) / T, T, 0, stream>>>(W2, w2t);

  const int strips = (N + 15) / 16;

  // --- layer 1: h1 = x@W1, o1 = b1 + dinv^2*h1 (fused) ; edge scatter ---
  k_gemm_wmma<H1, false><<<strips, 32 * (H1 / 16), 0, stream>>>(
      x, w1t, b1, dinv, h, o1, N);
  k_scatter<H1><<<(E + 7) / 8, T, 0, stream>>>(row, col, wn, h, o1, E);

  // --- layer 2: h2 = relu(o1)@W2, out = b2 + dinv^2*h2 (fused) ; scatter ---
  k_gemm_wmma<H2, true><<<strips, 32 * (H2 / 16), 0, stream>>>(
      o1, w2t, b2, dinv, h, out, N);
  k_scatter<H2><<<(E + 7) / 8, T, 0, stream>>>(row, col, wn, h, out, E);
}
